// MemoryAugmentedCNN_58789512347906
// MI455X (gfx1250) — compile-verified, matrix operand order checked by reading, not run on hardware
//
#include <hip/hip_runtime.h>
#include <math.h>

// MI455X / gfx1250, wave32.
// GEMM/conv: 128x64x32 workgroup tiles, 8 waves, each wave owns a 2x2 grid of
// 16x16 f32 accumulators fed by V_WMMA_F32_16X16X32_F16. A/B tiles are DMA'd
// into LDS (f32) by the Tensor Data Mover with a 2-deep double-buffered
// pipeline: tile i+1's TENSOR_LOAD_TO_LDS overlaps tile i's WMMAs, synced by
// S_WAIT_TENSORCNT 1 (tensor ops complete in-order per wave). f32->f16
// conversion happens during fragment packing (v_cvt_pk_f16_f32 co-executes
// with WMMA). Weights stream from HBM once; feats (52 MB) stays L2-resident.

typedef __attribute__((ext_vector_type(16))) _Float16 v16h;
typedef __attribute__((ext_vector_type(8)))  float    v8f;
typedef __attribute__((ext_vector_type(4)))  unsigned int u32x4;
typedef __attribute__((ext_vector_type(8)))  int      i32x8;
typedef __attribute__((ext_vector_type(4)))  int      i32x4;

#define BM 128
#define BN 64
#define BK 32
#define NT 256

#if __has_builtin(__builtin_amdgcn_tensor_load_to_lds) && __has_builtin(__builtin_amdgcn_s_wait_tensorcnt)
#define TDM_OK 1
#else
#define TDM_OK 0
#endif

// ---------------------------------------------------------------------------
// TDM: DMA a 2D f32 tile [tile_d1 rows x tile_d0 cols] from a strided tensor
// into LDS (rows packed contiguously). tensor_d0/d1 = remaining extents for
// OOB zero-fill. Wave-level op (EXEC ignored); completion via TENSORcnt.
// ---------------------------------------------------------------------------
#if TDM_OK
__device__ inline void tdm_load_2d_f32(unsigned lds_off, const float* gaddr,
                                       unsigned tensor_d0, unsigned tensor_d1,
                                       unsigned tile_d0, unsigned tile_d1,
                                       unsigned long long stride0_elems)
{
    unsigned long long ga = (unsigned long long)(uintptr_t)gaddr;
    u32x4 g0;
    g0[0] = 1u;                                            // count=1 (valid user D#)
    g0[1] = lds_off;                                       // lds_addr (bytes)
    g0[2] = (unsigned)(ga & 0xFFFFFFFFu);                  // global_addr[31:0]
    g0[3] = (unsigned)((ga >> 32) & 0x1FFFFFFu) | (2u << 30); // addr[56:32] | type=2
    i32x8 g1;
    g1[0] = (int)(2u << 16);                               // data_size=4B, wg_mask=0
    g1[1] = (int)((tensor_d0 & 0xFFFFu) << 16);            // tensor_dim0[15:0]
    g1[2] = (int)((tensor_d0 >> 16) | ((tensor_d1 & 0xFFFFu) << 16));
    g1[3] = (int)((tensor_d1 >> 16) | (tile_d0 << 16));    // tensor_dim1 hi | tile_dim0
    g1[4] = (int)(tile_d1 & 0xFFFFu);                      // tile_dim1, tile_dim2=0
    g1[5] = (int)(unsigned)(stride0_elems & 0xFFFFFFFFull);// tensor_dim0_stride lo
    g1[6] = (int)(unsigned)((stride0_elems >> 32) & 0xFFFFull);
    g1[7] = 0;
    i32x4 z4 = {0, 0, 0, 0};
#if defined(__clang_major__) && (__clang_major__ >= 23)
    i32x8 z8 = {0, 0, 0, 0, 0, 0, 0, 0};
    __builtin_amdgcn_tensor_load_to_lds(g0, g1, z4, z4, z8, 0);
#else
    __builtin_amdgcn_tensor_load_to_lds(g0, g1, z4, z4, 0);
#endif
}
#endif

// Pack one 16x16x32 f16 A/B fragment from an f32 LDS tile (row-major, ld=BK),
// converting f32->f16 per the CDNA5 16-bit operand VGPR layout.
__device__ inline v16h pack_frag_f32(const float* __restrict__ S, int rowBase32,
                                     int l15, int khi)
{
    union { v16h v; _Float16 h[16]; } f;
    const int base = (rowBase32 + l15) * BK;
    #pragma unroll
    for (int vv = 0; vv < 8; ++vv) {
        int kb = ((vv < 4) ? (2 * vv) : (16 + 2 * (vv - 4))) + khi;
        f.h[2 * vv]     = (_Float16)S[base + kb];
        f.h[2 * vv + 1] = (_Float16)S[base + kb + 1];
    }
    return f.v;
}

// ---------------------------------------------------------------------------
// GEMM: C[M,N] = A[M,K] * op(B) (+bias) (+C). op(B)=B^T if transB (B stored
// [N,K] row-major -> TDM tile), else B stored [K,N] (manual strided stage).
// REQUIRES: Mrows % 128 == 0, K % 32 == 0 (true for every call site).
// ---------------------------------------------------------------------------
__global__ __launch_bounds__(NT)
void gemm_wmma_kernel(const float* __restrict__ A, int lda,
                      const float* __restrict__ B, int ldb, int transB,
                      float* __restrict__ C, int ldc,
                      const float* __restrict__ bias, int accumulate,
                      int Mrows, int Ncols, int K)
{
    __shared__ float As32[2][BM * BK];   // 32 KB (double-buffered)
    __shared__ float Bs32[2][BN * BK];   // 16 KB (double-buffered)

    const int tid  = threadIdx.x;
    const int lane = tid & 31;
    const int wv   = tid >> 5;        // 0..7
    const int wm   = wv >> 1;         // 0..3 -> M base wm*32
    const int wn   = wv & 1;          // 0..1 -> N base wn*32
    const int l15  = lane & 15;
    const int khi  = (lane & 16) ? 8 : 0;
    const long rowBase = (long)blockIdx.x * BM;
    const long colBase = (long)blockIdx.y * BN;
    const int nIter = (K + BK - 1) / BK;

    v8f acc[2][2] = {};

#if TDM_OK
    // pipeline prologue: DMA tile 0
    if (tid == 0)
        tdm_load_2d_f32((unsigned)(uintptr_t)(void*)As32[0], A + rowBase * lda,
                        (unsigned)K, (unsigned)(Mrows - rowBase),
                        BK, BM, (unsigned long long)lda);
    if (transB && tid == 32)
        tdm_load_2d_f32((unsigned)(uintptr_t)(void*)Bs32[0], B + colBase * ldb,
                        (unsigned)K, (unsigned)(Ncols - colBase),
                        BK, BN, (unsigned long long)ldb);
#endif

    for (int it = 0; it < nIter; ++it) {
        const int k0  = it * BK;
        const int cur = it & 1;
        const int nxt = cur ^ 1;
        const bool more = (it + 1 < nIter);

#if TDM_OK
        // issue tile it+1 into the other buffer (its last readers finished
        // two barriers ago), then wait for tile it only (in-order completion)
        if (more) {
            int k1 = k0 + BK;
            if (tid == 0)
                tdm_load_2d_f32((unsigned)(uintptr_t)(void*)As32[nxt],
                                A + rowBase * lda + k1,
                                (unsigned)(K - k1), (unsigned)(Mrows - rowBase),
                                BK, BM, (unsigned long long)lda);
            if (transB && tid == 32)
                tdm_load_2d_f32((unsigned)(uintptr_t)(void*)Bs32[nxt],
                                B + colBase * ldb + k1,
                                (unsigned)(K - k1), (unsigned)(Ncols - colBase),
                                BK, BN, (unsigned long long)ldb);
        }
        if (more) __builtin_amdgcn_s_wait_tensorcnt(1);
        else      __builtin_amdgcn_s_wait_tensorcnt(0);
#else
        // manual A stage: 16 contiguous f32 per thread
        {
            int r = tid >> 1, kk0 = (tid & 1) * 16;
            const float* ap = A + (rowBase + r) * (long)lda + k0 + kk0;
            #pragma unroll
            for (int i = 0; i < 16; ++i) As32[cur][r * BK + kk0 + i] = ap[i];
        }
        if (transB) {
            #pragma unroll
            for (int i = 0; i < 8; ++i) {
                int idx = tid * 8 + i;
                int nn = idx >> 5, kk = idx & 31;
                long col = colBase + nn;
                Bs32[cur][nn * BK + kk] = (col < Ncols) ? B[col * (long)ldb + k0 + kk] : 0.0f;
            }
        }
#endif
        if (!transB) {
            // B stored [K,N]: strided gather (n fastest for coalescing)
            #pragma unroll
            for (int i = 0; i < 8; ++i) {
                int idx = i * NT + tid;
                int nn = idx & 63, kk = idx >> 6;
                long col = colBase + nn;
                float v = (col < Ncols) ? B[(long)(k0 + kk) * ldb + col] : 0.0f;
                Bs32[cur][nn * BK + kk] = v;
            }
        }
        __syncthreads();

        v16h a0 = pack_frag_f32(As32[cur], wm * 32,      l15, khi);
        v16h a1 = pack_frag_f32(As32[cur], wm * 32 + 16, l15, khi);
        v16h b0 = pack_frag_f32(Bs32[cur], wn * 32,      l15, khi);
        v16h b1 = pack_frag_f32(Bs32[cur], wn * 32 + 16, l15, khi);
        acc[0][0] = __builtin_amdgcn_wmma_f32_16x16x32_f16(false, a0, false, b0, (short)0, acc[0][0], false, false);
        acc[0][1] = __builtin_amdgcn_wmma_f32_16x16x32_f16(false, a0, false, b1, (short)0, acc[0][1], false, false);
        acc[1][0] = __builtin_amdgcn_wmma_f32_16x16x32_f16(false, a1, false, b0, (short)0, acc[1][0], false, false);
        acc[1][1] = __builtin_amdgcn_wmma_f32_16x16x32_f16(false, a1, false, b1, (short)0, acc[1][1], false, false);
        __syncthreads();
    }

    #pragma unroll
    for (int sn = 0; sn < 2; ++sn) {
        long col = colBase + wn * 32 + sn * 16 + l15;
        if (col < Ncols) {
            float bv = bias ? bias[col] : 0.0f;
            #pragma unroll
            for (int sm = 0; sm < 2; ++sm) {
                union { v8f v; float f[8]; } res; res.v = acc[sm][sn];
                #pragma unroll
                for (int vv = 0; vv < 8; ++vv) {
                    long mrow = rowBase + wm * 32 + sm * 16 + vv + ((lane & 16) ? 8 : 0);
                    long o = mrow * (long)ldc + col;
                    float out = res.f[vv] + bv;
                    if (accumulate) out += C[o];
                    C[o] = out;
                }
            }
        }
    }
}

// ---------------------------------------------------------------------------
// Implicit-GEMM conv (3x3 VALID) + bias. rows = n*OH*OW, cols = CO, K = CI*9.
// Weight tiles via double-buffered TDM; A tile = im2col gather (single LDS
// buffer is race-free: staging sits between the two barriers). REQUIRES
// Mrows % 128 == 0, CO % 64 == 0 (true for all 4 layers).
// ---------------------------------------------------------------------------
__global__ __launch_bounds__(NT)
void conv_wmma_kernel(const float* __restrict__ X, const float* __restrict__ Wt,
                      const float* __restrict__ bias, float* __restrict__ Y,
                      int Nimg, int CI, int IH, int IW, int CO, int OH, int OW)
{
    const int K = CI * 9;
    __shared__ float As32[BM * BK];      // 16 KB
    __shared__ float Bs32[2][BN * BK];   // 16 KB (double-buffered)

    const int tid  = threadIdx.x;
    const int lane = tid & 31;
    const int wv   = tid >> 5;
    const int wm   = wv >> 1;
    const int wn   = wv & 1;
    const int l15  = lane & 15;
    const int khi  = (lane & 16) ? 8 : 0;
    const int rowBase = blockIdx.x * BM;
    const int colBase = blockIdx.y * BN;
    const int HW = OH * OW;
    const int nIter = (K + BK - 1) / BK;

    v8f acc[2][2] = {};

#if TDM_OK
    if (tid == 0)
        tdm_load_2d_f32((unsigned)(uintptr_t)(void*)Bs32[0], Wt + colBase * K,
                        (unsigned)K, (unsigned)(CO - colBase),
                        BK, BN, (unsigned long long)K);
#endif

    for (int it = 0; it < nIter; ++it) {
        const int k0  = it * BK;
        const int cur = it & 1;
        const int nxt = cur ^ 1;
        const bool more = (it + 1 < nIter);

#if TDM_OK
        if (more && tid == 0) {
            int k1 = k0 + BK;
            tdm_load_2d_f32((unsigned)(uintptr_t)(void*)Bs32[nxt], Wt + colBase * K + k1,
                            (unsigned)(K - k1), (unsigned)(CO - colBase),
                            BK, BN, (unsigned long long)K);
        }
#else
        #pragma unroll
        for (int i = 0; i < 8; ++i) {
            int idx = tid * 8 + i;
            int nn = idx >> 5, kk = idx & 31;
            int k = k0 + kk;
            int kc = (k < K) ? k : 0;
            float v = Wt[(colBase + nn) * K + kc];
            Bs32[cur][nn * BK + kk] = (k < K) ? v : 0.0f;
        }
#endif
        // A tile: on-the-fly im2col gather, 16 elems/thread, branchless tail
        {
            int r = tid >> 1, kk0 = (tid & 1) * 16;
            int grow = rowBase + r;
            int n  = grow / HW;
            int p  = grow - n * HW;
            int oh = p / OW, ow = p - oh * OW;
            #pragma unroll
            for (int i = 0; i < 16; ++i) {
                int k = k0 + kk0 + i;
                int kc = (k < K) ? k : 0;
                int ci = kc / 9; int rem = kc - ci * 9;
                int kh = rem / 3, kw = rem - kh * 3;
                float v = X[((n * CI + ci) * IH + (oh + kh)) * IW + (ow + kw)];
                As32[r * BK + kk0 + i] = (k < K) ? v : 0.0f;
            }
        }
#if TDM_OK
        if (more) __builtin_amdgcn_s_wait_tensorcnt(1);
        else      __builtin_amdgcn_s_wait_tensorcnt(0);
#endif
        __syncthreads();

        v16h a0 = pack_frag_f32(As32,      wm * 32,      l15, khi);
        v16h a1 = pack_frag_f32(As32,      wm * 32 + 16, l15, khi);
        v16h b0 = pack_frag_f32(Bs32[cur], wn * 32,      l15, khi);
        v16h b1 = pack_frag_f32(Bs32[cur], wn * 32 + 16, l15, khi);
        acc[0][0] = __builtin_amdgcn_wmma_f32_16x16x32_f16(false, a0, false, b0, (short)0, acc[0][0], false, false);
        acc[0][1] = __builtin_amdgcn_wmma_f32_16x16x32_f16(false, a0, false, b1, (short)0, acc[0][1], false, false);
        acc[1][0] = __builtin_amdgcn_wmma_f32_16x16x32_f16(false, a1, false, b0, (short)0, acc[1][0], false, false);
        acc[1][1] = __builtin_amdgcn_wmma_f32_16x16x32_f16(false, a1, false, b1, (short)0, acc[1][1], false, false);
        __syncthreads();
    }

    #pragma unroll
    for (int sn = 0; sn < 2; ++sn) {
        int coN = colBase + wn * 32 + sn * 16 + l15;
        float bv = bias[coN];
        #pragma unroll
        for (int sm = 0; sm < 2; ++sm) {
            union { v8f v; float f[8]; } res; res.v = acc[sm][sn];
            #pragma unroll
            for (int vv = 0; vv < 8; ++vv) {
                int mrow = rowBase + wm * 32 + sm * 16 + vv + ((lane & 16) ? 8 : 0);
                int n = mrow / HW;
                int p = mrow - n * HW;
                Y[(n * CO + coN) * HW + p] = res.f[vv] + bv;
            }
        }
    }
}

// ---------------------------------------------------------------------------
// BatchNorm stats per channel (biased var): one block per channel.
// ---------------------------------------------------------------------------
__global__ __launch_bounds__(NT)
void bn_stats_kernel(const float* __restrict__ Y, float* __restrict__ mu,
                     float* __restrict__ rinv, int Nimg, int CO, int HW)
{
    const int c = blockIdx.x;
    const int cnt = Nimg * HW;
    float s = 0.0f, ss = 0.0f;
    for (int i = threadIdx.x; i < cnt; i += NT) {
        int n = i / HW, p = i - n * HW;
        float v = Y[(n * CO + c) * HW + p];
        s += v; ss += v * v;
    }
    __shared__ float sb[NT], ssb[NT];
    sb[threadIdx.x] = s; ssb[threadIdx.x] = ss;
    __syncthreads();
    for (int off = NT / 2; off > 0; off >>= 1) {
        if (threadIdx.x < off) { sb[threadIdx.x] += sb[threadIdx.x + off];
                                 ssb[threadIdx.x] += ssb[threadIdx.x + off]; }
        __syncthreads();
    }
    if (threadIdx.x == 0) {
        float m   = sb[0] / cnt;
        float var = ssb[0] / cnt - m * m;
        mu[c] = m;
        rinv[c] = rsqrtf(var + 1e-5f);
    }
}

__global__ __launch_bounds__(NT)
void bn_gelu_kernel(float* __restrict__ Y, const float* __restrict__ mu,
                    const float* __restrict__ rinv, const float* __restrict__ g,
                    const float* __restrict__ beta, int total, int CO, int HW)
{
    int idx = blockIdx.x * NT + threadIdx.x;
    if (idx >= total) return;
    int c = (idx / HW) % CO;
    float y = (Y[idx] - mu[c]) * rinv[c] * g[c] + beta[c];
    Y[idx] = 0.5f * y * (1.0f + erff(y * 0.70710678118654752f));  // exact GELU
}

// ---------------------------------------------------------------------------
// Row softmax in place, one block per row.
// ---------------------------------------------------------------------------
__global__ __launch_bounds__(NT)
void softmax_kernel(float* __restrict__ X, int cols)
{
    float* row = X + (long)blockIdx.x * cols;
    __shared__ float sb[NT];
    float mx = -3.402823466e38f;
    for (int i = threadIdx.x; i < cols; i += NT) mx = fmaxf(mx, row[i]);
    sb[threadIdx.x] = mx; __syncthreads();
    for (int off = NT / 2; off > 0; off >>= 1) {
        if (threadIdx.x < off) sb[threadIdx.x] = fmaxf(sb[threadIdx.x], sb[threadIdx.x + off]);
        __syncthreads();
    }
    mx = sb[0]; __syncthreads();
    float s = 0.0f;
    for (int i = threadIdx.x; i < cols; i += NT) {
        float e = expf(row[i] - mx);
        row[i] = e; s += e;
    }
    sb[threadIdx.x] = s; __syncthreads();
    for (int off = NT / 2; off > 0; off >>= 1) {
        if (threadIdx.x < off) sb[threadIdx.x] += sb[threadIdx.x + off];
        __syncthreads();
    }
    float inv = 1.0f / sb[0];
    for (int i = threadIdx.x; i < cols; i += NT) row[i] *= inv;
}

// ---------------------------------------------------------------------------
// Write path / usage / least-used tail
// ---------------------------------------------------------------------------
__global__ __launch_bounds__(NT)
void aww_kernel(const float* __restrict__ gate, const float* __restrict__ prw,
                const float* __restrict__ plu, float* __restrict__ aww,
                int Nb, int R, int T)
{
    int idx = blockIdx.x * NT + threadIdx.x;
    if (idx >= Nb * T) return;
    int n = idx / T, t = idx - n * T;
    float sg = 1.0f / (1.0f + expf(-gate[t]));
    float s = 0.0f;
    for (int r = 0; r < R; ++r) {
        long o = ((long)(n * R + r)) * T + t;
        s += sg * prw[o] + (1.0f - sg) * plu[o];
    }
    aww[idx] = s / (float)R;
}

__global__ __launch_bounds__(128)
void write_mem_kernel(const float* __restrict__ aww, const float* __restrict__ wk,
                      const float* __restrict__ mem, float* __restrict__ newmem,
                      int Nb, int T, int M)
{
    int t = blockIdx.x;
    int m = threadIdx.x;
    float acc = 0.0f;
    for (int n = 0; n < Nb; ++n)
        acc += aww[n * T + t] * wk[n * M + m];
    newmem[t * M + m] = mem[t * M + m] + acc;
}

__global__ __launch_bounds__(NT)
void usage_kernel(const float* __restrict__ prev_usage, const float* __restrict__ rw,
                  const float* __restrict__ gate, const float* __restrict__ prw,
                  const float* __restrict__ plu, float* __restrict__ usage,
                  int total, int T)
{
    int idx = blockIdx.x * NT + threadIdx.x;
    if (idx >= total) return;
    int t = idx % T;
    float sg  = 1.0f / (1.0f + expf(-gate[t]));
    float cww = sg * prw[idx] + (1.0f - sg) * plu[idx];
    usage[idx] = 0.95f * prev_usage[idx] + rw[idx] + cww;
}

// 4th-smallest per row (duplicate-safe): per-thread sorted-4 lists, LDS merge.
__global__ __launch_bounds__(NT)
void kth_kernel(const float* __restrict__ usage, float* __restrict__ kth, int T)
{
    const long r = blockIdx.x;
    float b[4] = {3.402823466e38f, 3.402823466e38f, 3.402823466e38f, 3.402823466e38f};
    for (int i = threadIdx.x; i < T; i += NT) {
        float u = usage[r * T + i];
        if (u < b[3]) {
            b[3] = u;
            for (int j = 3; j > 0 && b[j] < b[j - 1]; --j) {
                float tmp = b[j]; b[j] = b[j - 1]; b[j - 1] = tmp;
            }
        }
    }
    __shared__ float sb[NT * 4];
    #pragma unroll
    for (int j = 0; j < 4; ++j) sb[threadIdx.x * 4 + j] = b[j];
    __syncthreads();
    for (int off = NT / 2; off > 0; off >>= 1) {
        if (threadIdx.x < off) {
            float* a = &sb[threadIdx.x * 4];
            float* c = &sb[(threadIdx.x + off) * 4];
            float o[4]; int ia = 0, ib = 0;
            #pragma unroll
            for (int i = 0; i < 4; ++i) o[i] = (a[ia] <= c[ib]) ? a[ia++] : c[ib++];
            #pragma unroll
            for (int i = 0; i < 4; ++i) a[i] = o[i];
        }
        __syncthreads();
    }
    if (threadIdx.x == 0) kth[r] = sb[3];
}

__global__ __launch_bounds__(NT)
void least_used_kernel(const float* __restrict__ usage, const float* __restrict__ kth,
                       float* __restrict__ lu, int total, int T)
{
    int idx = blockIdx.x * NT + threadIdx.x;
    if (idx >= total) return;
    int row = idx / T;
    lu[idx] = (usage[idx] <= kth[row]) ? 1.0f : 0.0f;
}

// ---------------------------------------------------------------------------
// Host-side orchestration
// ---------------------------------------------------------------------------
extern "C" void kernel_launch(void* const* d_in, const int* in_sizes, int n_in,
                              void* d_out, int out_size, void* d_ws, size_t ws_size,
                              hipStream_t stream)
{
    (void)in_sizes; (void)n_in; (void)out_size; (void)ws_size;

    // setup_inputs() insertion order: inputs, conv[l]{W,b,g,beta} x4,
    // Wr,br,Ww,bw,Wo,bo, external_memory, sigmoid_gate, prev_read_weights,
    // prev_usage_weights, prev_least_used_weights.
    const float* x_in = (const float*)d_in[0];
    const float* cW[4]; const float* cb[4]; const float* cg[4]; const float* cbe[4];
    for (int l = 0; l < 4; ++l) {
        cW[l]  = (const float*)d_in[1 + 4 * l + 0];
        cb[l]  = (const float*)d_in[1 + 4 * l + 1];
        cg[l]  = (const float*)d_in[1 + 4 * l + 2];
        cbe[l] = (const float*)d_in[1 + 4 * l + 3];
    }
    const float* Wr   = (const float*)d_in[17];
    const float* br   = (const float*)d_in[18];
    const float* Ww   = (const float*)d_in[19];
    const float* bw   = (const float*)d_in[20];
    const float* Wo   = (const float*)d_in[21];
    const float* bo   = (const float*)d_in[22];
    const float* mem  = (const float*)d_in[23];
    const float* gate = (const float*)d_in[24];
    const float* prw  = (const float*)d_in[25];
    const float* pus  = (const float*)d_in[26];
    const float* plu  = (const float*)d_in[27];

    const int Nb = 256, R = 4, Mm = 128, T = 4096;
    const int FEAT = 51200, NC = 1000;

    // workspace layout (floats); ~157 MB total
    float* ws   = (float*)d_ws;
    float* actA = ws;
    float* actB = actA + (size_t)16 * 1024 * 1024;
    float* mu   = actB + (size_t)16 * 1024 * 1024;
    float* rinv = mu + 128;
    float* rk   = rinv + 128;                    // [256,512] read keys
    float* wk   = rk + 256 * 512;                // [256,128] write keys
    float* att  = wk + 256 * 128;                // [1024,4096] scores/read_weights
    float* rv   = att + (size_t)1024 * 4096;     // [1024,128] read vectors
    float* aww  = rv + 1024 * 128;               // [256,4096]
    float* kthv = aww + (size_t)256 * 4096;      // [1024]

    // output segments (return order)
    float* logits = (float*)d_out;               // [256,1000]
    float* newmem = logits + 256 * NC;           // [4096,128]
    float* usage  = newmem + (size_t)T * Mm;     // [256,4,4096]
    float* lu     = usage + (size_t)Nb * R * T;  // [256,4,4096]

    // ---- conv + BN + GELU chain ----
    const int ci[4] = {3, 64, 64, 128};
    const int co[4] = {64, 64, 128, 128};
    const int ih[4] = {28, 26, 24, 22};
    const int oh[4] = {26, 24, 22, 20};
    float* bufs[2] = {actA, actB};
    const float* cur = x_in;
    for (int l = 0; l < 4; ++l) {
        float* outb = bufs[l & 1];
        int HW = oh[l] * oh[l];
        int Mrows = Nb * HW;                     // divisible by 128 for all layers
        dim3 g(Mrows / BM, co[l] / BN);
        conv_wmma_kernel<<<g, NT, 0, stream>>>(cur, cW[l], cb[l], outb,
                                               Nb, ci[l], ih[l], ih[l], co[l], oh[l], oh[l]);
        bn_stats_kernel<<<co[l], NT, 0, stream>>>(outb, mu, rinv, Nb, co[l], HW);
        int total = Nb * co[l] * HW;
        bn_gelu_kernel<<<(total + NT - 1) / NT, NT, 0, stream>>>(outb, mu, rinv,
                                                                 cg[l], cbe[l], total, co[l], HW);
        cur = outb;
    }
    const float* feats = cur;  // [256, 51200] = actB

    // ---- controller GEMMs ----
    gemm_wmma_kernel<<<dim3(2, 8), NT, 0, stream>>>(feats, FEAT, Wr, FEAT, 1,
                                                    rk, 512, br, 0, 256, 512, FEAT);
    gemm_wmma_kernel<<<dim3(2, 2), NT, 0, stream>>>(feats, FEAT, Ww, FEAT, 1,
                                                    wk, 128, bw, 0, 256, 128, FEAT);

    // ---- read: scores, softmax, read vectors ----
    gemm_wmma_kernel<<<dim3(8, 64), NT, 0, stream>>>(rk, Mm, mem, Mm, 1,
                                                     att, T, nullptr, 0, 1024, T, Mm);
    softmax_kernel<<<1024, NT, 0, stream>>>(att, T);
    gemm_wmma_kernel<<<dim3(8, 2), NT, 0, stream>>>(att, T, mem, Mm, 0,
                                                    rv, Mm, nullptr, 0, 1024, Mm, T);

    // ---- output head: logits = feats@Wo[:, :FEAT]^T + rv@Wo[:, FEAT:]^T + bo ----
    dim3 gl(2, (NC + BN - 1) / BN);
    gemm_wmma_kernel<<<gl, NT, 0, stream>>>(feats, FEAT, Wo, FEAT + R * Mm, 1,
                                            logits, NC, bo, 0, 256, NC, FEAT);
    gemm_wmma_kernel<<<gl, NT, 0, stream>>>(rv, R * Mm, Wo + FEAT, FEAT + R * Mm, 1,
                                            logits, NC, nullptr, 1, 256, NC, R * Mm);

    // ---- write path / usage / least-used ----
    aww_kernel<<<(Nb * T + NT - 1) / NT, NT, 0, stream>>>(gate, prw, plu, aww, Nb, R, T);
    write_mem_kernel<<<T, Mm, 0, stream>>>(aww, wk, mem, newmem, Nb, T, Mm);
    int tot = Nb * R * T;
    usage_kernel<<<(tot + NT - 1) / NT, NT, 0, stream>>>(pus, att, gate, prw, plu,
                                                         usage, tot, T);
    kth_kernel<<<Nb * R, NT, 0, stream>>>(usage, kthv, T);
    least_used_kernel<<<(tot + NT - 1) / NT, NT, 0, stream>>>(usage, kthv, lu, tot, T);
}
